// COCOSpeaker_9912784519289
// MI455X (gfx1250) — compile-verified, matrix operand order checked by reading, not run on hardware
//
#include <hip/hip_runtime.h>
#include <math.h>

// ---------------- constants ----------------
#define B_   1024
#define T_   20
#define D_   512
#define V_   10000
#define VP_  10048   // V padded to multiple of 64 for 16x64 wave slabs
#define H_   64
#define FI_  2048
#define FB_  256
#define KIN_ 2304
#define G3_  1536

typedef unsigned short u16;
typedef __attribute__((ext_vector_type(16))) __bf16 v16bf;
typedef __attribute__((ext_vector_type(8)))  float  v8f;
typedef __attribute__((ext_vector_type(4)))  unsigned int u32x4;

__device__ __forceinline__ u16 f2bf(float x) {
    unsigned u = __builtin_bit_cast(unsigned, x);
    u += 0x7FFFu + ((u >> 16) & 1u);           // round-to-nearest-even
    return (u16)(u >> 16);
}
__device__ __forceinline__ float bf2f(u16 h) {
    return __builtin_bit_cast(float, ((unsigned)h) << 16);
}

union FragU { v16bf v; u32x4 q[2]; };

// A fragment (16x32 bf16) per ISA layout: lane L holds row M=L%16, half=L/16,
// elements = K[8*half .. 8*half+7] then K[16+8*half .. +7]  -> two b128 loads.
__device__ __forceinline__ v16bf load_frag(const u16* __restrict__ base, int ld,
                                           int row, int k, int half) {
    const u16* p = base + (long)row * ld + k + 8 * half;
    FragU f;
    f.q[0] = *(const u32x4*)(p);
    f.q[1] = *(const u32x4*)(p + 16);
    return f.v;
}

// B fragment from an LDS slab laid out [64 rows(n)][32 cols(k)] bf16 (64B rows).
__device__ __forceinline__ v16bf lds_frag(const u16* sb, int j, int l, int half) {
    const u16* p = sb + (((j << 4) + l) << 5) + 8 * half;
    FragU f;
    f.q[0] = *(const u32x4*)(p);        // -> ds_load_b128
    f.q[1] = *(const u32x4*)(p + 16);
    return f.v;
}

__device__ __forceinline__ void wait_asynccnt0() {
#if __has_builtin(__builtin_amdgcn_s_wait_asynccnt)
    __builtin_amdgcn_s_wait_asynccnt(0);
#else
    asm volatile("s_wait_asynccnt 0" ::: "memory");
#endif
}

// ---------------- WMMA GEMM with async-to-LDS double-buffered B staging -------
// Block = 256 threads = 8 waves, computes a 128(M) x 64(N) supertile.
// Wave w owns rows [tm*128 + 16w, +16) and all 4 N-subtiles (4 accumulators).
// B slab (64 rows x 32 k, 4KB) is staged global->LDS with
// GLOBAL_LOAD_ASYNC_TO_LDS_B128 (ASYNCcnt) and double-buffered across K-chunks.
// C[M,N](f32) = epi( A[M,K](bf16 rm) x WT[N,K](bf16 rm)^T + bias[N] )
// EPI: 0 none, 1 exact GELU, 2 tanh. Optional bf16 mirror Cbf.
template <int EPI>
__global__ void gemm_bf16_wmma(const u16* __restrict__ A, const u16* __restrict__ WT,
                               const float* __restrict__ bias,
                               float* __restrict__ C, u16* __restrict__ Cbf,
                               int M, int N, int K) {
    __shared__ __align__(16) u16 sB[2][64 * 32];

    int tilesN64 = N >> 6;
    int tm = blockIdx.x / tilesN64;
    int tn = blockIdx.x - tm * tilesN64;
    int tid  = threadIdx.x;
    int w    = tid >> 5;
    int lane = tid & 31;
    int half = lane >> 4, l = lane & 15;
    int arow  = (tm << 7) + (w << 4) + l;
    int nbase = tn << 6;
    int nk = K >> 5;

    // stage B slab for K-chunk kc into buffer buf: 256 lanes x 16B = 4KB
    int srow = tid >> 2;                 // 0..63  (n within slab)
    int scol = (tid & 3) << 3;           // 0,8,16,24 (k element)
    auto stageB = [&](int kc, int buf) {
        const u16* gp = WT + (long)(nbase + srow) * K + (kc << 5) + scol;
        unsigned ldsoff = (unsigned)(size_t)(void*)(&sB[buf][0]) + (unsigned)(tid << 4);
        unsigned long long gaddr = (unsigned long long)(size_t)(const void*)gp;
        asm volatile("global_load_async_to_lds_b128 %0, %1, off"
                     :: "v"(ldsoff), "v"(gaddr) : "memory");
    };

    stageB(0, 0);
    wait_asynccnt0();
    __syncthreads();

    v8f acc[4] = {};
    for (int kc = 0; kc < nk; ++kc) {
        int buf = kc & 1;
        if (kc + 1 < nk) {
            stageB(kc + 1, buf ^ 1);                         // overlap with compute
            __builtin_prefetch((const void*)(A + (long)arow * K + ((kc + 1) << 5)), 0, 1);
        }
        v16bf a = load_frag(A, K, arow, kc << 5, half);
        const u16* sb = &sB[buf][0];
        v16bf b0 = lds_frag(sb, 0, l, half);
        v16bf b1 = lds_frag(sb, 1, l, half);
        v16bf b2 = lds_frag(sb, 2, l, half);
        v16bf b3 = lds_frag(sb, 3, l, half);
        acc[0] = __builtin_amdgcn_wmma_f32_16x16x32_bf16(false, a, false, b0, (short)0, acc[0], false, false);
        acc[1] = __builtin_amdgcn_wmma_f32_16x16x32_bf16(false, a, false, b1, (short)0, acc[1], false, false);
        acc[2] = __builtin_amdgcn_wmma_f32_16x16x32_bf16(false, a, false, b2, (short)0, acc[2], false, false);
        acc[3] = __builtin_amdgcn_wmma_f32_16x16x32_bf16(false, a, false, b3, (short)0, acc[3], false, false);
        if (kc + 1 < nk) {
            wait_asynccnt0();     // next buffer landed in LDS
            __syncthreads();      // + everyone done reading current buffer
        }
    }

#pragma unroll
    for (int j = 0; j < 4; ++j) {
        int col = nbase + (j << 4) + l;
        float bv = bias ? bias[col] : 0.0f;
#pragma unroll
        for (int r = 0; r < 8; ++r) {
            int m = (tm << 7) + (w << 4) + r + 8 * half;
            float v = acc[j][r] + bv;
            if (EPI == 1) v = 0.5f * v * (1.0f + erff(v * 0.70710678118654752f));
            else if (EPI == 2) v = tanhf(v);
            long o = (long)m * N + col;
            if (C)   C[o]   = v;
            if (Cbf) Cbf[o] = f2bf(v);
        }
    }
}

// ---------------- weight transpose + f32->bf16 ----------------
__global__ void transpose_cvt_kernel(const float* __restrict__ src, u16* __restrict__ dst,
                                     int K, int N) {
    long idx = (long)blockIdx.x * blockDim.x + threadIdx.x;
    if (idx < (long)K * N) {
        int k = (int)(idx / N), n = (int)(idx - (long)k * N);
        dst[(long)n * K + k] = f2bf(src[idx]);
    }
}

__global__ void cvt_inputs_kernel(const float* __restrict__ img, const float* __restrict__ box,
                                  u16* __restrict__ dst) {
    long idx = (long)blockIdx.x * blockDim.x + threadIdx.x;
    if (idx < (long)B_ * KIN_) {
        int row = (int)(idx / KIN_), c = (int)(idx - (long)row * KIN_);
        float v = (c < FI_) ? img[(long)row * FI_ + c] : box[(long)row * FB_ + (c - FI_)];
        dst[idx] = f2bf(v);
    }
}

__global__ void zero_ix_kernel(float* __restrict__ ix, u16* __restrict__ ixbf) {
    long idx = (long)blockIdx.x * blockDim.x + threadIdx.x;
    if (idx < (long)B_ * D_) { ix[idx] = 0.0f; ixbf[idx] = f2bf(0.0f); }
}

__global__ void addmask_kernel(const float* __restrict__ b, const float* __restrict__ m,
                               float* __restrict__ dst) {
    int i = blockIdx.x * blockDim.x + threadIdx.x;
    if (i < VP_) dst[i] = (i < V_) ? (b[i] + m[i]) : -1000.0f;  // pad cols heavily masked
}

// ---------------- LayerNorm: out_bf = LN(a (+res)) (*g+b) ; block=512, grid=B ----
__global__ void ln_kernel(const float* __restrict__ a, const float* __restrict__ res,
                          const float* __restrict__ g, const float* __restrict__ bta,
                          u16* __restrict__ obf) {
    __shared__ float ssum[D_], ssq[D_];
    int row = blockIdx.x, c = threadIdx.x;
    long o = (long)row * D_ + c;
    float v = a[o] + (res ? res[o] : 0.0f);
    ssum[c] = v; ssq[c] = v * v;
    __syncthreads();
    for (int s = D_ / 2; s > 0; s >>= 1) {
        if (c < s) { ssum[c] += ssum[c + s]; ssq[c] += ssq[c + s]; }
        __syncthreads();
    }
    float mean = ssum[0] * (1.0f / D_);
    float var  = ssq[0] * (1.0f / D_) - mean * mean;
    float y = (v - mean) * rsqrtf(var + 1e-5f);
    if (g) y = y * g[c] + bta[c];
    obf[o] = f2bf(y);
}

// ---------------- GRU gates: h = (1-z)*n + z*ix (torch r,z,n order) -------------
__global__ void gru_gate_kernel(const float* __restrict__ gi, const float* __restrict__ gh,
                                const float* __restrict__ ix,
                                float* __restrict__ hx, u16* __restrict__ hxbf) {
    long idx = (long)blockIdx.x * blockDim.x + threadIdx.x;
    if (idx >= (long)B_ * D_) return;
    int row = (int)(idx >> 9), c = (int)(idx & (D_ - 1));
    long b3 = (long)row * G3_;
    float gir = gi[b3 + c], giz = gi[b3 + D_ + c], gin = gi[b3 + 2 * D_ + c];
    float ghr = gh[b3 + c], ghz = gh[b3 + D_ + c], ghn = gh[b3 + 2 * D_ + c];
    float r = 1.0f / (1.0f + expf(-(gir + ghr)));
    float z = 1.0f / (1.0f + expf(-(giz + ghz)));
    float n = tanhf(gin + r * ghn);
    float h = (1.0f - z) * n + z * ix[idx];
    hx[idx] = h; hxbf[idx] = f2bf(h);
}

// ---------------- categorical sampling (Gumbel argmax) + logp + entropy --------
__device__ __forceinline__ unsigned hash3(unsigned a, unsigned b, unsigned c) {
    unsigned h = a * 747796405u + b * 2891336453u + c * 2654435769u + 0x9E3779B9u;
    h ^= h >> 16; h *= 0x7feb352du; h ^= h >> 15; h *= 0x846ca68bu; h ^= h >> 16;
    return h;
}
__global__ void sample_kernel(const float* __restrict__ logits, float* __restrict__ out,
                              int* __restrict__ actsInt, int t) {
    __shared__ float sm[256], sc[256];
    __shared__ int   si[256];
    int row = blockIdx.x, tid = threadIdx.x;
    const float* lrow = logits + (long)row * VP_;
    float lmax = -3.0e38f, bscore = -3.0e38f; int bidx = 0;
    for (int v = tid; v < V_; v += 256) {
        float lg = lrow[v];
        lmax = fmaxf(lmax, lg);
        unsigned h = hash3(0x1234567u + (unsigned)t, (unsigned)row, (unsigned)v);
        float u = (float)(h >> 9) * 1.1920929e-7f + 1e-12f;
        float gmb = -logf(-logf(u));
        float s = lg + gmb;
        if (s > bscore) { bscore = s; bidx = v; }
    }
    sm[tid] = lmax; sc[tid] = bscore; si[tid] = bidx;
    __syncthreads();
    for (int s = 128; s > 0; s >>= 1) {
        if (tid < s) {
            sm[tid] = fmaxf(sm[tid], sm[tid + s]);
            if (sc[tid + s] > sc[tid]) { sc[tid] = sc[tid + s]; si[tid] = si[tid + s]; }
        }
        __syncthreads();
    }
    float m = sm[0]; int act = si[0];
    __syncthreads();
    float se = 0.0f, sl = 0.0f;
    for (int v = tid; v < V_; v += 256) {
        float e = expf(lrow[v] - m);
        se += e; sl += e * lrow[v];
    }
    sm[tid] = se; sc[tid] = sl;
    __syncthreads();
    for (int s = 128; s > 0; s >>= 1) {
        if (tid < s) { sm[tid] += sm[tid + s]; sc[tid] += sc[tid + s]; }
        __syncthreads();
    }
    if (tid == 0) {
        float seT = sm[0], slT = sc[0];
        float lse = m + logf(seT);
        out[(long)row * T_ + t]                  = (float)act;        // acts
        out[(long)B_ * T_ + (long)row * T_ + t]  = lrow[act] - lse;   // logp
        out[2L * B_ * T_ + (long)row * T_ + t]   = lse - slT / seT;   // entropy
        actsInt[row] = act;
    }
}

// ---------------- critic final dot: vals[b,t] = ch2 . c3_w + c3_b --------------
__global__ void critic_out_kernel(const u16* __restrict__ ch2bf, const float* __restrict__ c3w,
                                  const float* __restrict__ c3b, float* __restrict__ out, int t) {
    int row = blockIdx.x * blockDim.x + threadIdx.x;
    if (row < B_) {
        float s = 0.0f;
#pragma unroll
        for (int i = 0; i < H_; ++i) s += bf2f(ch2bf[(long)row * H_ + i]) * c3w[i];
        out[3L * B_ * T_ + (long)row * T_ + t] = s + c3b[0];
    }
}

// ---------------- embedding gather: ix = emb[act] ------------------------------
__global__ void gather_emb_kernel(const int* __restrict__ actsInt, const float* __restrict__ emb,
                                  float* __restrict__ ix, u16* __restrict__ ixbf) {
    long idx = (long)blockIdx.x * blockDim.x + threadIdx.x;
    if (idx >= (long)B_ * D_) return;
    int row = (int)(idx >> 9), c = (int)(idx & (D_ - 1));
    float v = emb[(long)actsInt[row] * D_ + c];
    ix[idx] = v; ixbf[idx] = f2bf(v);
}

// ---------------- host launcher ----------------
extern "C" void kernel_launch(void* const* d_in, const int* in_sizes, int n_in,
                              void* d_out, int out_size, void* d_ws, size_t ws_size,
                              hipStream_t stream) {
    const float* image = (const float*)d_in[0];
    const float* box   = (const float*)d_in[1];
    const float* fr_w  = (const float*)d_in[2];  const float* fr_b  = (const float*)d_in[3];
    const float* d1a_w = (const float*)d_in[4];  const float* d1a_b = (const float*)d_in[5];
    const float* d1b_w = (const float*)d_in[6];  const float* d1b_b = (const float*)d_in[7];
    const float* d2a_w = (const float*)d_in[8];  const float* d2a_b = (const float*)d_in[9];
    const float* d2b_w = (const float*)d_in[10]; const float* d2b_b = (const float*)d_in[11];
    const float* ln_g  = (const float*)d_in[12]; const float* ln_b  = (const float*)d_in[13];
    const float* eo_w  = (const float*)d_in[14]; const float* eo_b  = (const float*)d_in[15];
    const float* emb   = (const float*)d_in[16];
    const float* gwi   = (const float*)d_in[17]; const float* gwh   = (const float*)d_in[18];
    const float* gbi   = (const float*)d_in[19]; const float* gbh   = (const float*)d_in[20];
    const float* a1_w  = (const float*)d_in[21]; const float* a1_b  = (const float*)d_in[22];
    const float* a2_w  = (const float*)d_in[23]; const float* a2_b  = (const float*)d_in[24];
    const float* a3_w  = (const float*)d_in[25]; const float* a3_b  = (const float*)d_in[26];
    const float* c1_w  = (const float*)d_in[27]; const float* c1_b  = (const float*)d_in[28];
    const float* c2_w  = (const float*)d_in[29]; const float* c2_b  = (const float*)d_in[30];
    const float* c3_w  = (const float*)d_in[31]; const float* c3_b  = (const float*)d_in[32];
    const float* wmask = (const float*)d_in[33];
    float* out = (float*)d_out;

    // ---- workspace carve (256B aligned) ----
    char* base = (char*)d_ws;
    size_t off = 0;
    auto carve = [&](size_t bytes) -> char* {
        off = (off + 255) & ~(size_t)255;
        char* p = base + off; off += bytes; return p;
    };
    u16* actbuf = (u16*)carve((size_t)B_ * KIN_ * 2);
    u16* frT    = (u16*)carve((size_t)D_ * KIN_ * 2);
    u16* d1aT   = (u16*)carve((size_t)D_ * D_ * 2);
    u16* d1bT   = (u16*)carve((size_t)D_ * D_ * 2);
    u16* d2aT   = (u16*)carve((size_t)D_ * D_ * 2);
    u16* d2bT   = (u16*)carve((size_t)D_ * D_ * 2);
    u16* eoT    = (u16*)carve((size_t)D_ * D_ * 2);
    u16* gwiT   = (u16*)carve((size_t)G3_ * D_ * 2);
    u16* gwhT   = (u16*)carve((size_t)G3_ * D_ * 2);
    u16* a1T    = (u16*)carve((size_t)H_ * D_ * 2);
    u16* a2T    = (u16*)carve((size_t)H_ * H_ * 2);
    u16* a3T    = (u16*)carve((size_t)VP_ * H_ * 2);   // padded rows
    u16* c1T    = (u16*)carve((size_t)H_ * D_ * 2);
    u16* c2T    = (u16*)carve((size_t)H_ * H_ * 2);
    float* biasMask = (float*)carve((size_t)VP_ * 4);
    float* x0     = (float*)carve((size_t)B_ * D_ * 4);
    float* bufA   = (float*)carve((size_t)B_ * D_ * 4);
    float* gi     = (float*)carve((size_t)B_ * G3_ * 4);
    float* gh     = (float*)carve((size_t)B_ * G3_ * 4);
    float* logits = (float*)carve((size_t)B_ * VP_ * 4);
    float* hx     = (float*)carve((size_t)B_ * D_ * 4);
    float* ix     = (float*)carve((size_t)B_ * D_ * 4);
    u16* ln0bf = (u16*)carve((size_t)B_ * D_ * 2);
    u16* t1bf  = (u16*)carve((size_t)B_ * D_ * 2);
    u16* ybf   = (u16*)carve((size_t)B_ * D_ * 2);
    u16* t2bf  = (u16*)carve((size_t)B_ * D_ * 2);
    u16* lnxbf = (u16*)carve((size_t)B_ * D_ * 2);
    u16* hxbf  = (u16*)carve((size_t)B_ * D_ * 2);
    u16* ixbf  = (u16*)carve((size_t)B_ * D_ * 2);
    u16* ah1bf = (u16*)carve((size_t)B_ * H_ * 2);
    u16* ah2bf = (u16*)carve((size_t)B_ * H_ * 2);
    u16* ch1bf = (u16*)carve((size_t)B_ * H_ * 2);
    u16* ch2bf = (u16*)carve((size_t)B_ * H_ * 2);
    int* actsInt = (int*)carve((size_t)B_ * 4);
    (void)ws_size; (void)in_sizes; (void)n_in; (void)out_size;

    auto tc = [&](const float* s, u16* d, int K, int N) {
        long n = (long)K * N;
        transpose_cvt_kernel<<<(unsigned)((n + 255) / 256), 256, 0, stream>>>(s, d, K, N);
    };
    auto gemm = [&](int epi, const u16* A, const u16* WT, const float* bias,
                    float* C, u16* Cbf, int M, int N, int K) {
        int blocks = (M / 128) * (N / 64);
        if (epi == 0)      gemm_bf16_wmma<0><<<blocks, 256, 0, stream>>>(A, WT, bias, C, Cbf, M, N, K);
        else if (epi == 1) gemm_bf16_wmma<1><<<blocks, 256, 0, stream>>>(A, WT, bias, C, Cbf, M, N, K);
        else               gemm_bf16_wmma<2><<<blocks, 256, 0, stream>>>(A, WT, bias, C, Cbf, M, N, K);
    };

    // ---- per-call conversions ----
    tc(fr_w,  frT,  KIN_, D_);
    tc(d1a_w, d1aT, D_, D_);  tc(d1b_w, d1bT, D_, D_);
    tc(d2a_w, d2aT, D_, D_);  tc(d2b_w, d2bT, D_, D_);
    tc(eo_w,  eoT,  D_, D_);
    tc(gwi,   gwiT, D_, G3_); tc(gwh,   gwhT, D_, G3_);
    tc(a1_w,  a1T,  D_, H_);  tc(a2_w,  a2T,  H_, H_);  tc(a3_w, a3T, H_, V_);
    tc(c1_w,  c1T,  D_, H_);  tc(c2_w,  c2T,  H_, H_);
    addmask_kernel<<<(VP_ + 255) / 256, 256, 0, stream>>>(a3_b, wmask, biasMask);
    cvt_inputs_kernel<<<(B_ * KIN_ + 255) / 256, 256, 0, stream>>>(image, box, actbuf);
    zero_ix_kernel<<<(B_ * D_ + 255) / 256, 256, 0, stream>>>(ix, ixbf);

    // ---- encoder ----
    gemm(0, actbuf, frT, fr_b, x0, nullptr, B_, D_, KIN_);
    ln_kernel<<<B_, D_, 0, stream>>>(x0, nullptr, nullptr, nullptr, ln0bf);
    gemm(1, ln0bf, d1aT, d1a_b, nullptr, t1bf, B_, D_, D_);
    gemm(0, t1bf,  d1bT, d1b_b, bufA, nullptr, B_, D_, D_);
    ln_kernel<<<B_, D_, 0, stream>>>(bufA, x0, nullptr, nullptr, ybf);
    gemm(1, ybf,  d2aT, d2a_b, nullptr, t2bf, B_, D_, D_);
    gemm(0, t2bf, d2bT, d2b_b, bufA, nullptr, B_, D_, D_);
    ln_kernel<<<B_, D_, 0, stream>>>(bufA, x0, ln_g, ln_b, lnxbf);
    gemm(0, lnxbf, eoT, eo_b, hx, hxbf, B_, D_, D_);   // img -> initial hx (+bf16)

    // ---- GRU rollout ----
    for (int t = 0; t < T_; ++t) {
        gemm(0, hxbf, gwiT, gbi, gi, nullptr, B_, G3_, D_);
        gemm(0, ixbf, gwhT, gbh, gh, nullptr, B_, G3_, D_);
        gru_gate_kernel<<<(B_ * D_ + 255) / 256, 256, 0, stream>>>(gi, gh, ix, hx, hxbf);
        // actor
        gemm(2, hxbf,  a1T, a1_b, nullptr, ah1bf, B_, H_, D_);
        gemm(2, ah1bf, a2T, a2_b, nullptr, ah2bf, B_, H_, H_);
        gemm(0, ah2bf, a3T, biasMask, logits, nullptr, B_, VP_, H_);
        // critic
        gemm(2, hxbf,  c1T, c1_b, nullptr, ch1bf, B_, H_, D_);
        gemm(2, ch1bf, c2T, c2_b, nullptr, ch2bf, B_, H_, H_);
        critic_out_kernel<<<(B_ + 255) / 256, 256, 0, stream>>>(ch2bf, c3_w, c3_b, out, t);
        // sample + next-step embedding input
        sample_kernel<<<B_, 256, 0, stream>>>(logits, out, actsInt, t);
        gather_emb_kernel<<<(B_ * D_ + 255) / 256, 256, 0, stream>>>(actsInt, emb, ix, ixbf);
    }
}
